// Predict_imu_39539468927110
// MI455X (gfx1250) — matrix operand order repaired and checked by reading, not compiled, not run on hardware
//
#include <hip/hip_runtime.h>
#include <hip/hip_bf16.h>

// ---------------------------------------------------------------------------
// Types for CDNA5 WMMA (wave32): bf16 operands, fp32 accumulate.
// ---------------------------------------------------------------------------
typedef __bf16 bf16;
typedef __attribute__((ext_vector_type(16))) __bf16 v16bf;
typedef __attribute__((ext_vector_type(8)))  __bf16 bf16x8;
typedef __attribute__((ext_vector_type(8)))  float  v8f;

#define EW_BLOCK 256
static inline unsigned gridFor(size_t n) { return (unsigned)((n + EW_BLOCK - 1) / EW_BLOCK); }

// ---------------------------------------------------------------------------
// Fragment loaders. Operand VGPR layouts per CDNA5 ISA 7.12.2:
//   A: lane<16 -> row m, halves = K {0..7, 16..23};  lane>=16 -> K {8..15, 24..31}
//   B: lane<16 -> col n, halves = K {0..7, 16..23};  lane>=16 -> K {8..15, 24..31}
// ---------------------------------------------------------------------------
__device__ __forceinline__ v16bf frag_a(const bf16* __restrict__ row, int k0, int ka0) {
  bf16x8 lo = *(const bf16x8*)(row + k0 + ka0);
  bf16x8 hi = *(const bf16x8*)(row + k0 + 16 + ka0);
  v16bf v;
#pragma unroll
  for (int i = 0; i < 8; ++i) { v[i] = lo[i]; v[i + 8] = hi[i]; }
  return v;
}
__device__ __forceinline__ v16bf frag_b(const bf16* __restrict__ row, int k0, int kb0) {
  bf16x8 lo = *(const bf16x8*)(row + k0 + kb0);
  bf16x8 hi = *(const bf16x8*)(row + k0 + 16 + kb0);
  v16bf v;
#pragma unroll
  for (int i = 0; i < 8; ++i) { v[i] = lo[i]; v[i + 8] = hi[i]; }
  return v;
}

// ---------------------------------------------------------------------------
// WMMA GEMM:  C[M,N] = A[M,K] * BT[N,K]^T  (+bias[n]) (+relu)
// A, BT bf16, zero-padded to 32-granular rows / 32-granular K so all tile
// loads are unguarded & 16B aligned. Each wave computes a 32x32 C region
// (2x2 register-blocked 16x16 tiles -> 4 WMMAs per 8 fragment loads);
// 8 waves per 256-thread block.
//   C: vgpr r, lane<16 -> M = m+r, lane>=16 -> M = m+r+8; N = n+(lane&15)
// ---------------------------------------------------------------------------
__global__ void k_wmma_gemm(const bf16* __restrict__ A, int lda,
                            const bf16* __restrict__ BT, int ldb,
                            float* __restrict__ C, int ldc,
                            bf16* __restrict__ Cbf, int ldcb,
                            const float* __restrict__ bias,
                            int M, int N, int K, int Mt, int Nt, int relu)
{
  int wave = blockIdx.x * (blockDim.x >> 5) + (threadIdx.x >> 5);
  if (wave >= Mt * Nt) return;                 // wave-uniform: EXEC all-1s for WMMA
  int lane = threadIdx.x & 31;
  int l15  = lane & 15;
  int half = lane >> 4;

  int tm = wave / Nt, tn = wave % Nt;
  int m0 = tm << 5,  n0 = tn << 5;

  const bf16* a0 = A  + (size_t)(m0 + l15) * lda;
  const bf16* a1 = a0 + (size_t)16 * lda;
  const bf16* b0 = BT + (size_t)(n0 + l15) * ldb;
  const bf16* b1 = b0 + (size_t)16 * ldb;
  int ka0 = half * 8;    // A lane K offset
  int kb0 = half * 8;    // B lane K offset

  v8f acc[2][2] = {};
  for (int k0 = 0; k0 < K; k0 += 32) {
    if (k0 + 64 <= K) {                        // prefetch next K chunk
      __builtin_prefetch(a0 + k0 + 32, 0, 1);
      __builtin_prefetch(a1 + k0 + 32, 0, 1);
      __builtin_prefetch(b0 + k0 + 32, 0, 1);
      __builtin_prefetch(b1 + k0 + 32, 0, 1);
    }
    v16bf av0 = frag_a(a0, k0, ka0);
    v16bf av1 = frag_a(a1, k0, ka0);
    v16bf bv0 = frag_b(b0, k0, kb0);
    v16bf bv1 = frag_b(b1, k0, kb0);
    acc[0][0] = __builtin_amdgcn_wmma_f32_16x16x32_bf16(false, av0, false, bv0,
                                                        (short)0, acc[0][0], false, false);
    acc[0][1] = __builtin_amdgcn_wmma_f32_16x16x32_bf16(false, av0, false, bv1,
                                                        (short)0, acc[0][1], false, false);
    acc[1][0] = __builtin_amdgcn_wmma_f32_16x16x32_bf16(false, av1, false, bv0,
                                                        (short)0, acc[1][0], false, false);
    acc[1][1] = __builtin_amdgcn_wmma_f32_16x16x32_bf16(false, av1, false, bv1,
                                                        (short)0, acc[1][1], false, false);
  }

#pragma unroll
  for (int ni = 0; ni < 2; ++ni) {
    int col = n0 + ni * 16 + l15;
    float bval = (bias && col < N) ? bias[col] : 0.0f;
#pragma unroll
    for (int mi = 0; mi < 2; ++mi) {
#pragma unroll
      for (int r = 0; r < 8; ++r) {
        int row = m0 + mi * 16 + r + half * 8;
        float v = acc[mi][ni][r] + bval;
        if (relu) v = fmaxf(v, 0.0f);
        bool in = (row < M) && (col < N);
        if (C && in) C[(size_t)row * ldc + col] = v;
        if (Cbf) Cbf[(size_t)row * ldcb + col] = in ? (bf16)v : (bf16)0.0f;
      }
    }
  }
}

// ---------------------------------------------------------------------------
// Elementwise / staging kernels
// ---------------------------------------------------------------------------
__global__ void k_zero_bf(bf16* p, int n) {
  int i = blockIdx.x * blockDim.x + threadIdx.x;
  if (i < n) p[i] = (bf16)0.0f;
}
__global__ void k_zero_f(float* p, int n) {
  int i = blockIdx.x * blockDim.x + threadIdx.x;
  if (i < n) p[i] = 0.0f;
}

// fp32 (rows x cols, row-major, contiguous) -> zero-padded bf16 (rpad x ldd)
__global__ void k_cvtpad(const float* __restrict__ src, bf16* __restrict__ dst,
                         int rows, int cols, int ldd, int total) {
  int i = blockIdx.x * blockDim.x + threadIdx.x;
  if (i >= total) return;
  int c = i % ldd, r = i / ldd;
  dst[i] = (r < rows && c < cols) ? (bf16)src[(size_t)r * cols + c] : (bf16)0.0f;
}

__device__ __forceinline__ float sigf(float x) { return 1.0f / (1.0f + __expf(-x)); }

// LSTM cell epilogue (gate order i,f,g,o). h written as bf16 into the concat
// activation buffer that is the A operand of the next GEMM.
__global__ void k_gates(const float* __restrict__ xg, const float* __restrict__ hp,
                        const float* __restrict__ bih, const float* __restrict__ bhh,
                        float* __restrict__ c, bf16* __restrict__ hout, int ldh,
                        int B, int H) {
  int idx = blockIdx.x * blockDim.x + threadIdx.x;
  if (idx >= B * H) return;
  int b = idx / H, j = idx % H;
  const float* xr = xg + (size_t)b * 4 * H;
  const float* hr = hp + (size_t)b * 4 * H;
  float gi = xr[j]         + hr[j]         + bih[j]         + bhh[j];
  float gf = xr[H + j]     + hr[H + j]     + bih[H + j]     + bhh[H + j];
  float gg = xr[2 * H + j] + hr[2 * H + j] + bih[2 * H + j] + bhh[2 * H + j];
  float go = xr[3 * H + j] + hr[3 * H + j] + bih[3 * H + j] + bhh[3 * H + j];
  float cv = sigf(gf) * c[idx] + sigf(gi) * tanhf(gg);
  c[idx] = cv;
  hout[(size_t)b * ldh + j] = (bf16)(sigf(go) * tanhf(cv));
}

// BatchNorm1d (training-mode batch stats, biased var) over (B, nfeat) contiguous,
// optional ReLU and post-ReLU residual add. One thread per feature.
__global__ void k_bn(const float* __restrict__ in, float* __restrict__ out,
                     const float* __restrict__ g, const float* __restrict__ b,
                     int nfeat, int B, int relu, const float* __restrict__ resid) {
  int j = blockIdx.x * blockDim.x + threadIdx.x;
  if (j >= nfeat) return;
  float s = 0.0f, s2 = 0.0f;
  for (int i = 0; i < B; ++i) { float v = in[(size_t)i * nfeat + j]; s += v; s2 += v * v; }
  float inv = 1.0f / (float)B;
  float mu = s * inv;
  float var = s2 * inv - mu * mu;
  float sc = g[j] * rsqrtf(var + 1e-5f);
  float bb = b[j] - mu * sc;
  for (int i = 0; i < B; ++i) {
    float v = in[(size_t)i * nfeat + j] * sc + bb;
    if (relu) v = fmaxf(v, 0.0f);
    if (resid) v += resid[(size_t)i * nfeat + j];
    out[(size_t)i * nfeat + j] = v;
  }
}

// Graph node-mix: y[b,f,m*3+d] = sum_n x[b,f,n*3+d] * (Ap*M+Q)[n,m]
__global__ void k_nodemix(const float* __restrict__ x, int sB, int sF,
                          const float* __restrict__ Ap, const float* __restrict__ Mm,
                          const float* __restrict__ Q, float* __restrict__ y,
                          int B, int F, int node) {
  int idx = blockIdx.x * blockDim.x + threadIdx.x;
  int total = B * F * node * 3;
  if (idx >= total) return;
  int dd = idx % 3;
  int m  = (idx / 3) % node;
  int f  = (idx / (3 * node)) % F;
  int b  = idx / (3 * node * F);
  float s = 0.0f;
  for (int n2 = 0; n2 < node; ++n2) {
    float att = Ap[n2 * node + m] * Mm[n2 * node + m] + Q[n2 * node + m];
    s += x[(size_t)b * sB + (size_t)f * sF + n2 * 3 + dd] * att;
  }
  y[((size_t)b * F + f) * (node * 3) + m * 3 + dd] = s;
}

// permute y(b,g,n) -> BT[(b*n) x Kp] bf16 for the frame-mix GEMM (W as A operand)
__global__ void k_perm_bt(const float* __restrict__ y, bf16* __restrict__ bt,
                          int B, int F, int n, int Kp) {
  int idx = blockIdx.x * blockDim.x + threadIdx.x;
  int total = B * n * Kp;
  if (idx >= total) return;
  int g = idx % Kp, col = idx / Kp;
  int nn = col % n, b = col / n;
  bt[idx] = (g < F) ? (bf16)y[((size_t)b * F + g) * n + nn] : (bf16)0.0f;
}

// z[b,f,nn] = Ct[f, b*n+nn] + bias[nn]
__global__ void k_unperm(const float* __restrict__ Ct, const float* __restrict__ bias,
                         float* __restrict__ z, int B, int F, int n) {
  int idx = blockIdx.x * blockDim.x + threadIdx.x;
  int total = B * F * n;
  if (idx >= total) return;
  int nn = idx % n, f = (idx / n) % F, b = idx / (n * F);
  z[idx] = Ct[(size_t)f * (B * n) + (size_t)b * n + nn] + bias[nn];
}

// (F,B,C) -> (B,F,C)
__global__ void k_tr_fbc(const float* __restrict__ in, float* __restrict__ out,
                         int F, int B, int C) {
  int idx = blockIdx.x * blockDim.x + threadIdx.x;
  int total = F * B * C;
  if (idx >= total) return;
  int c = idx % C, b = (idx / C) % B, f = idx / (C * B);
  out[((size_t)b * F + f) * C + c] = in[idx];
}

// Up-block concat: out(b,f,0..c-1)=y ; out(b,f,c..c+2)= y·up_w^T + up_b
__global__ void k_upcat(const float* __restrict__ y, const float* __restrict__ w,
                        const float* __restrict__ ub, float* __restrict__ out,
                        int B, int F, int c) {
  int co = c + 3;
  int idx = blockIdx.x * blockDim.x + threadIdx.x;
  int total = B * F * co;
  if (idx >= total) return;
  int cc = idx % co, bf = idx / co;
  if (cc < c) { out[idx] = y[(size_t)bf * c + cc]; return; }
  int r = cc - c;
  const float* yr = y + (size_t)bf * c;
  float s = ub[r];
  for (int k = 0; k < c; ++k) s += yr[k] * w[r * c + k];
  out[idx] = s;
}

// all_pos input: A[(s*B+b), k] = k<72 ? x[s,b,k] : k<102 ? y_phys[b,s,k-72] : 0
__global__ void k_cat_cvt(const float* __restrict__ x, const float* __restrict__ y,
                          bf16* __restrict__ A, int Kp, int S, int B) {
  int idx = blockIdx.x * blockDim.x + threadIdx.x;
  int total = S * B * Kp;
  if (idx >= total) return;
  int k = idx % Kp, sb = idx / Kp;
  int b = sb % B, s = sb / B;
  float v = 0.0f;
  if (k < 72)       v = x[((size_t)s * B + b) * 72 + k];
  else if (k < 102) v = y[((size_t)b * S + s) * 30 + (k - 72)];
  A[idx] = (bf16)v;
}

// ---------------------------------------------------------------------------
// Host-side helpers
// ---------------------------------------------------------------------------
struct Bump { char* base; size_t off; };
static void* balloc(Bump& a, size_t bytes) {
  void* p = a.base + a.off;
  a.off = (a.off + bytes + 255) & ~(size_t)255;
  return p;
}

static void gemm(hipStream_t st, const bf16* A, int lda, const bf16* BT, int ldb,
                 float* C, int ldc, bf16* Cbf, int ldcb, const float* bias,
                 int M, int N, int K, int relu) {
  int Mt = (M + 31) >> 5, Nt = (N + 31) >> 5;       // 32x32 per wave (2x2 blocked)
  unsigned blocks = (unsigned)(((long)Mt * Nt + 7) / 8);
  k_wmma_gemm<<<blocks, 256, 0, st>>>(A, lda, BT, ldb, C, ldc, Cbf, ldcb, bias,
                                      M, N, K, Mt, Nt, relu);
}
static void cvtpad(hipStream_t st, const float* src, bf16* dst,
                   int rows, int cols, int ldd, int rpad) {
  int total = rpad * ldd;
  k_cvtpad<<<gridFor((size_t)total), EW_BLOCK, 0, st>>>(src, dst, rows, cols, ldd, total);
}
static void bn(hipStream_t st, const float* in, float* out, const float* g,
               const float* b, int nfeat, int B, int relu, const float* resid) {
  k_bn<<<gridFor((size_t)nfeat), EW_BLOCK, 0, st>>>(in, out, g, b, nfeat, B, relu, resid);
}

// Parameter views (insertion-order DFS over the reference params dict)
struct DirW { const float *wih, *whh, *bih, *bhh; };
struct BiLW { const float *fc1w, *fc1b, *fc2w, *fc2b; DirW d[2][2]; int H, I, O, Kp; };
struct GcW  { const float *W, *Ap, *M, *Q, *b; };
struct GcbW { GcW g1, g2; const float *bn1g, *bn1b, *bn2g, *bn2b; };
struct UpW  { const float *bn1g, *bn1b, *upw, *upb; GcW gl; const float *bn2g, *bn2b; };

struct LstmWs {
  bf16 *h0, *l1, *l2, *hzero, *wihbf, *whhbf, *fc1wbf, *fc2wbf;
  float *xg, *hproj, *cbuf;
};

static void run_bilstm(hipStream_t st, const BiLW& P, const bf16* Ain,
                       float* outfc2, const LstmWs& w, int S, int B) {
  const int H = P.H, G = 4 * H, SB = S * B;
  // fc1 -> relu, bf16 output = layer-0 input
  cvtpad(st, P.fc1w, w.fc1wbf, H, P.I, P.Kp, H);
  gemm(st, Ain, P.Kp, w.fc1wbf, P.Kp, nullptr, 0, w.h0, H, P.fc1b, SB, H, P.Kp, 1);

  const bf16* lin = w.h0;
  int lda = H, Kl = H;
  bf16* louts[2] = { w.l1, w.l2 };
  for (int layer = 0; layer < 2; ++layer) {
    bf16* lout = louts[layer];               // (SB x 2H) bf16 concat
    for (int dir = 0; dir < 2; ++dir) {
      const DirW& dw = P.d[layer][dir];
      cvtpad(st, dw.wih, w.wihbf, G, Kl, Kl, G);
      cvtpad(st, dw.whh, w.whhbf, G, H, H, G);
      // input projection for all timesteps at once
      gemm(st, lin, lda, w.wihbf, Kl, w.xg, G, nullptr, 0, nullptr, SB, G, Kl, 0);
      k_zero_f<<<gridFor((size_t)B * H), EW_BLOCK, 0, st>>>(w.cbuf, B * H);
      for (int s = 0; s < S; ++s) {
        int t = (dir == 0) ? s : (S - 1 - s);
        int pt = (dir == 0) ? t - 1 : t + 1;
        const bf16* hA = (s == 0) ? w.hzero
                                  : lout + (size_t)pt * B * (2 * H) + dir * H;
        gemm(st, hA, 2 * H, w.whhbf, H, w.hproj, G, nullptr, 0, nullptr, B, G, H, 0);
        k_gates<<<gridFor((size_t)B * H), EW_BLOCK, 0, st>>>(
            w.xg + (size_t)t * B * G, w.hproj, dw.bih, dw.bhh, w.cbuf,
            lout + (size_t)t * B * (2 * H) + dir * H, 2 * H, B, H);
      }
    }
    lin = lout; lda = 2 * H; Kl = 2 * H;
  }
  // fc2 (N-pad 32-granular for the 32x32 wave tiles)
  int Op = ((P.O + 31) / 32) * 32;
  cvtpad(st, P.fc2w, w.fc2wbf, P.O, 2 * H, 2 * H, Op);
  gemm(st, lin, 2 * H, w.fc2wbf, 2 * H, outfc2, P.O, nullptr, 0, P.fc2b,
       SB, P.O, 2 * H, 0);
}

static void gconv(hipStream_t st, const float* xin, int sB, int sF, const GcW& p,
                  float* z, int B, int F, int n,
                  float* ybuf, bf16* Wbf, bf16* permBT, float* Cmat) {
  int node = n / 3;
  k_nodemix<<<gridFor((size_t)B * F * n), EW_BLOCK, 0, st>>>(
      xin, sB, sF, p.Ap, p.M, p.Q, ybuf, B, F, node);
  cvtpad(st, p.W, Wbf, F, F, 128, 128);
  k_perm_bt<<<gridFor((size_t)B * n * 128), EW_BLOCK, 0, st>>>(ybuf, permBT, B, F, n, 128);
  gemm(st, Wbf, 128, permBT, 128, Cmat, B * n, nullptr, 0, nullptr, F, B * n, 128, 0);
  k_unperm<<<gridFor((size_t)B * F * n), EW_BLOCK, 0, st>>>(Cmat, p.b, z, B, F, n);
}

static void gcblock(hipStream_t st, const GcbW& p, const float* xin, float* out,
                    int B, int F, int n, float* t1, float* t2,
                    float* ybuf, bf16* Wbf, bf16* permBT, float* Cmat) {
  gconv(st, xin, F * n, n, p.g1, t1, B, F, n, ybuf, Wbf, permBT, Cmat);
  bn(st, t1, t2, p.bn1g, p.bn1b, F * n, B, 1, nullptr);
  gconv(st, t2, F * n, n, p.g2, t1, B, F, n, ybuf, Wbf, permBT, Cmat);
  bn(st, t1, out, p.bn2g, p.bn2b, F * n, B, 1, xin);
}

static void upblock(hipStream_t st, const UpW& p, const float* xin, float* out,
                    int B, int F, int cin, float* t1, float* t2,
                    float* ybuf, bf16* Wbf, bf16* permBT, float* Cmat) {
  int cout = cin + 3;
  bn(st, xin, t1, p.bn1g, p.bn1b, F * cin, B, 0, nullptr);
  k_upcat<<<gridFor((size_t)B * F * cout), EW_BLOCK, 0, st>>>(t1, p.upw, p.upb, t2, B, F, cin);
  gconv(st, t2, F * cout, cout, p.gl, t1, B, F, cout, ybuf, Wbf, permBT, Cmat);
  bn(st, t1, out, p.bn2g, p.bn2b, F * cout, B, 1, t2);
}

// ---------------------------------------------------------------------------
// kernel_launch
// ---------------------------------------------------------------------------
extern "C" void kernel_launch(void* const* d_in, const int* in_sizes, int n_in,
                              void* d_out, int out_size, void* d_ws, size_t ws_size,
                              hipStream_t stream) {
  (void)in_sizes; (void)n_in; (void)out_size; (void)ws_size;
  const int S = 120, B = 512, SB = S * B;
  int ii = 0;
  auto nx = [&]() { return (const float*)d_in[ii++]; };
  const float* x = nx();

  auto rdBil = [&](int H, int I, int O, int Kp) {
    BiLW m; m.H = H; m.I = I; m.O = O; m.Kp = Kp;
    m.fc1w = nx(); m.fc1b = nx(); m.fc2w = nx(); m.fc2b = nx();
    for (int l = 0; l < 2; ++l)
      for (int dd = 0; dd < 2; ++dd) {
        m.d[l][dd].wih = nx(); m.d[l][dd].whh = nx();
        m.d[l][dd].bih = nx(); m.d[l][dd].bhh = nx();
      }
    return m;
  };
  auto rdGc  = [&]() { GcW g; g.W = nx(); g.Ap = nx(); g.M = nx(); g.Q = nx(); g.b = nx(); return g; };
  auto rdGcb = [&]() { GcbW g; g.g1 = rdGc(); g.g2 = rdGc();
                       g.bn1g = nx(); g.bn1b = nx(); g.bn2g = nx(); g.bn2b = nx(); return g; };
  auto rdUp  = [&]() { UpW u; u.bn1g = nx(); u.bn1b = nx(); u.upw = nx(); u.upb = nx();
                       u.gl = rdGc(); u.bn2g = nx(); u.bn2b = nx(); return u; };

  BiLW rnn  = rdBil(256, 72, 18, 96);
  GcbW gcn1 = rdGcb();
  UpW  ups[4]; for (int i = 0; i < 4; ++i) ups[i] = rdUp();
  BiLW allp = rdBil(128, 102, 72, 128);
  GcbW gcn2 = rdGcb();

  // workspace layout
  Bump a { (char*)d_ws, 0 };
  bf16*  A1     = (bf16*) balloc(a, (size_t)SB * 96  * 2);
  bf16*  A2     = (bf16*) balloc(a, (size_t)SB * 128 * 2);
  LstmWs lw;
  lw.h0    = (bf16*) balloc(a, (size_t)SB * 256 * 2);
  lw.l1    = (bf16*) balloc(a, (size_t)SB * 512 * 2);
  lw.l2    = (bf16*) balloc(a, (size_t)SB * 512 * 2);
  lw.hzero = (bf16*) balloc(a, (size_t)B  * 512 * 2);
  lw.wihbf = (bf16*) balloc(a, (size_t)1024 * 512 * 2);
  lw.whhbf = (bf16*) balloc(a, (size_t)1024 * 256 * 2);
  lw.fc1wbf= (bf16*) balloc(a, (size_t)256 * 128 * 2);
  lw.fc2wbf= (bf16*) balloc(a, (size_t)96  * 512 * 2);
  lw.xg    = (float*)balloc(a, (size_t)SB * 1024 * 4);
  lw.hproj = (float*)balloc(a, (size_t)B  * 1024 * 4);
  lw.cbuf  = (float*)balloc(a, (size_t)B  * 256  * 4);
  float* pos  = (float*)balloc(a, (size_t)SB * 18 * 4);
  float* out2 = (float*)balloc(a, (size_t)SB * 72 * 4);
  size_t GBUF = (size_t)B * S * 80;
  float* bufA = (float*)balloc(a, GBUF * 4);
  float* bufB = (float*)balloc(a, GBUF * 4);
  float* bufC = (float*)balloc(a, GBUF * 4);
  float* bufD = (float*)balloc(a, GBUF * 4);
  float* bufE = (float*)balloc(a, GBUF * 4);
  bf16*  Wbf    = (bf16*) balloc(a, (size_t)128 * 128 * 2);
  bf16*  permBT = (bf16*) balloc(a, (size_t)B * 72 * 128 * 2);
  float* Cmat   = (float*)balloc(a, (size_t)128 * B * 72 * 4);

  k_zero_bf<<<gridFor((size_t)B * 512), EW_BLOCK, 0, stream>>>(lw.hzero, B * 512);

  // ---- rnn_pos BiLSTM: x -> pos (S,B,18) ----
  cvtpad(stream, x, A1, SB, 72, 96, SB);
  run_bilstm(stream, rnn, A1, pos, lw, S, B);

  // ---- gcn1 on (B,F,18) ----
  k_tr_fbc<<<gridFor((size_t)SB * 18), EW_BLOCK, 0, stream>>>(pos, bufA, S, B, 18);
  gcblock(stream, gcn1, bufA, bufB, B, S, 18, bufC, bufD, bufE, Wbf, permBT, Cmat);

  // ---- progressive Up blocks: 18 -> 21 -> 24 -> 27 -> 30 (physical (B,F,c)) ----
  float* cur = bufB; float* nxt2 = bufA; int c = 18;
  for (int i = 0; i < 4; ++i) {
    upblock(stream, ups[i], cur, nxt2, B, S, c, bufC, bufD, bufE, Wbf, permBT, Cmat);
    float* t = cur; cur = nxt2; nxt2 = t; c += 3;
  }

  // ---- all_pos BiLSTM on concat([x, y], dim=2) ----
  k_cat_cvt<<<gridFor((size_t)SB * 128), EW_BLOCK, 0, stream>>>(x, cur, A2, 128, S, B);
  run_bilstm(stream, allp, A2, out2, lw, S, B);

  // ---- gcn2 on (B,F,72) -> d_out ----
  k_tr_fbc<<<gridFor((size_t)SB * 72), EW_BLOCK, 0, stream>>>(out2, bufA, S, B, 72);
  gcblock(stream, gcn2, bufA, (float*)d_out, B, S, 72, bufC, bufD, bufE, Wbf, permBT, Cmat);
}
